// AudioCodebook_2044404433533
// MI455X (gfx1250) — matrix operand order repaired and checked by reading, not compiled, not run on hardware
//
#include <hip/hip_runtime.h>
#include <math.h>
#include <stdint.h>

typedef __attribute__((ext_vector_type(16))) __bf16 v16bf;
typedef __attribute__((ext_vector_type(8)))  __bf16 v8bf;
typedef __attribute__((ext_vector_type(4)))  __bf16 v4bf;
typedef __attribute__((ext_vector_type(8)))  float  v8f;

#define F_DIM   32000
#define BATCH   512
#define KCODES  1024
#define MT      64
#define NT      64
#define KC      64
#define KPAD    72   // padded row stride (bf16): 144 B = 36 dwords -> conflict-free b128

// Workspace layout (floats):
//   [0, 512)                : x2
//   [512, 1536)             : c2
//   [1536, 1536+512*1024)   : d2 [B, K]
//   next 512 ints           : best index per row
//   (fast path only) bf16 planes: Lhi, Llo [512*32000], Chi, Clo [1024*32000]
#define WS_D2_OFF   (BATCH + KCODES)
#define WS_IDX_OFF  (WS_D2_OFF + BATCH * KCODES)
#define WS_SPLIT_OFF ((size_t)(WS_IDX_OFF + 512))          // 526336 floats, 16B aligned
#define WS_NEED_FAST (WS_SPLIT_OFF * 4 + (size_t)(2 * BATCH + 2 * KCODES) * F_DIM * 2)

// ---------------------------------------------------------------------------
__device__ __forceinline__ void async_b128(uint32_t lds_addr,
                                           const __bf16* sbase,
                                           uint32_t voff) {
    // GLOBAL_LOAD_ASYNC_TO_LDS_B128, GVS mode: mem = SADDR + VADDR32 + IOFFSET
    asm volatile("global_load_async_to_lds_b128 %0, %1, %2"
                 :: "v"(lds_addr), "v"(voff), "s"(sbase)
                 : "memory");
}
__device__ __forceinline__ void wait_async0() {
    asm volatile("s_wait_asynccnt 0x0" ::: "memory");
}

// ---------------------------------------------------------------------------
// Kernel 1 (slow path): row squared norms only
// ---------------------------------------------------------------------------
__global__ __launch_bounds__(256) void vq_norms(const float* __restrict__ lat,
                                                const float* __restrict__ cb,
                                                float* __restrict__ ws) {
    int r = blockIdx.x;
    const float* src = (r < BATCH) ? (lat + (size_t)r * F_DIM)
                                   : (cb + (size_t)(r - BATCH) * F_DIM);
    float s = 0.f;
    for (int i = threadIdx.x; i < F_DIM; i += 256) {
        float v = src[i];
        s += v * v;
    }
    #pragma unroll
    for (int off = 16; off > 0; off >>= 1) s += __shfl_down(s, off, 32);
    __shared__ float part[8];
    int lane = threadIdx.x & 31, w = threadIdx.x >> 5;
    if (lane == 0) part[w] = s;
    __syncthreads();
    if (threadIdx.x == 0) {
        float t = 0.f;
        #pragma unroll
        for (int i = 0; i < 8; ++i) t += part[i];
        ws[r] = t;
    }
}

// ---------------------------------------------------------------------------
// Kernel 1 (fast path): norms + one-time f32 -> (hi,lo) bf16 plane split
// ---------------------------------------------------------------------------
__global__ __launch_bounds__(256) void vq_norms_split(const float* __restrict__ lat,
                                                      const float* __restrict__ cb,
                                                      float* __restrict__ ws,
                                                      __bf16* __restrict__ Lhi,
                                                      __bf16* __restrict__ Llo,
                                                      __bf16* __restrict__ Chi,
                                                      __bf16* __restrict__ Clo) {
    int r = blockIdx.x;
    bool isL = (r < BATCH);
    size_t row = isL ? (size_t)r : (size_t)(r - BATCH);
    const float4* src = (const float4*)((isL ? lat : cb) + row * F_DIM);
    __bf16* hi = (isL ? Lhi : Chi) + row * F_DIM;
    __bf16* lo = (isL ? Llo : Clo) + row * F_DIM;

    float s = 0.f;
    for (int i = threadIdx.x; i < F_DIM / 4; i += 256) {
        float4 v = src[i];
        s += v.x * v.x + v.y * v.y + v.z * v.z + v.w * v.w;
        __bf16 h0 = (__bf16)v.x, h1 = (__bf16)v.y,
               h2 = (__bf16)v.z, h3 = (__bf16)v.w;
        v4bf hi4 = {h0, h1, h2, h3};
        v4bf lo4 = {(__bf16)(v.x - (float)h0), (__bf16)(v.y - (float)h1),
                    (__bf16)(v.z - (float)h2), (__bf16)(v.w - (float)h3)};
        *(v4bf*)(hi + i * 4) = hi4;
        *(v4bf*)(lo + i * 4) = lo4;
    }
    #pragma unroll
    for (int off = 16; off > 0; off >>= 1) s += __shfl_down(s, off, 32);
    __shared__ float part[8];
    int lane = threadIdx.x & 31, w = threadIdx.x >> 5;
    if (lane == 0) part[w] = s;
    __syncthreads();
    if (threadIdx.x == 0) {
        float t = 0.f;
        #pragma unroll
        for (int i = 0; i < 8; ++i) t += part[i];
        ws[r] = t;
    }
}

// ---------------------------------------------------------------------------
// Kernel 2 (fast path): GEMM from pre-split bf16 planes; async LDS staging.
// ---------------------------------------------------------------------------
__global__ __launch_bounds__(256) void vq_gemm_d2_bf16(const __bf16* __restrict__ Lhi,
                                                       const __bf16* __restrict__ Llo,
                                                       const __bf16* __restrict__ Chi,
                                                       const __bf16* __restrict__ Clo,
                                                       float* __restrict__ ws) {
    const float* x2 = ws;
    const float* c2 = ws + BATCH;
    float* d2 = ws + WS_D2_OFF;

    __shared__ __bf16 sAhi[2][MT * KPAD];
    __shared__ __bf16 sAlo[2][MT * KPAD];
    __shared__ __bf16 sBhi[2][NT * KPAD];
    __shared__ __bf16 sBlo[2][NT * KPAD];

    const int b0 = blockIdx.x * MT;
    const int n0 = blockIdx.y * NT;

    const int tid   = threadIdx.x;
    const int lane  = tid & 31;
    const int wv    = tid >> 5;
    const int mw    = wv >> 1;      // 0..3
    const int nw    = wv & 1;       // 0..1
    const int lhalf = lane >> 4;
    const int l16   = lane & 15;

    // Per-thread staging slots: tile = 64 rows x 128 B (per plane) = 512 b128;
    // 256 threads x 2 slots. Same (row,cv) reused for all 4 planes.
    uint32_t voff[2], ldso[2];
    #pragma unroll
    for (int j = 0; j < 2; ++j) {
        int id  = tid + j * 256;     // 0..511
        int row = id >> 3;           // 0..63
        int cv  = id & 7;            // 16-byte column
        voff[j] = (uint32_t)(row * (F_DIM * 2) + cv * 16);
        ldso[j] = (uint32_t)(row * (KPAD * 2) + cv * 16);
    }
    const __bf16* aHi = Lhi + (size_t)b0 * F_DIM;
    const __bf16* aLo = Llo + (size_t)b0 * F_DIM;
    const __bf16* bHi = Chi + (size_t)n0 * F_DIM;
    const __bf16* bLo = Clo + (size_t)n0 * F_DIM;

    const uint32_t lAhi0 = (uint32_t)(size_t)&sAhi[0][0];
    const uint32_t lAhi1 = (uint32_t)(size_t)&sAhi[1][0];
    const uint32_t lAlo0 = (uint32_t)(size_t)&sAlo[0][0];
    const uint32_t lAlo1 = (uint32_t)(size_t)&sAlo[1][0];
    const uint32_t lBhi0 = (uint32_t)(size_t)&sBhi[0][0];
    const uint32_t lBhi1 = (uint32_t)(size_t)&sBhi[1][0];
    const uint32_t lBlo0 = (uint32_t)(size_t)&sBlo[0][0];
    const uint32_t lBlo1 = (uint32_t)(size_t)&sBlo[1][0];

    v8f acc0 = {};
    v8f acc1 = {};

    // Prologue: stage chunk 0 into buffer 0.
    #pragma unroll
    for (int j = 0; j < 2; ++j) {
        async_b128(lAhi0 + ldso[j], aHi, voff[j]);
        async_b128(lAlo0 + ldso[j], aLo, voff[j]);
        async_b128(lBhi0 + ldso[j], bHi, voff[j]);
        async_b128(lBlo0 + ldso[j], bLo, voff[j]);
    }

    const int aRow = (mw * 16 + l16) * KPAD;

    for (int f0 = 0, it = 0; f0 < F_DIM; f0 += KC, ++it) {
        const int buf = it & 1;
        wait_async0();          // current buffer's copies done (this wave)
        __syncthreads();        // ... and every wave's

        int fn = f0 + KC;
        if (fn < F_DIM) {       // stage next chunk into the other buffer
            const uint32_t nAhi = buf ? lAhi0 : lAhi1;
            const uint32_t nAlo = buf ? lAlo0 : lAlo1;
            const uint32_t nBhi = buf ? lBhi0 : lBhi1;
            const uint32_t nBlo = buf ? lBlo0 : lBlo1;
            #pragma unroll
            for (int j = 0; j < 2; ++j) {
                async_b128(nAhi + ldso[j], aHi + fn, voff[j]);
                async_b128(nAlo + ldso[j], aLo + fn, voff[j]);
                async_b128(nBhi + ldso[j], bHi + fn, voff[j]);
                async_b128(nBlo + ldso[j], bLo + fn, voff[j]);
            }
        }

        const __bf16* AhiR = (buf ? sAhi[1] : sAhi[0]) + aRow;
        const __bf16* AloR = (buf ? sAlo[1] : sAlo[0]) + aRow;
        const __bf16* Bh   = (buf ? sBhi[1] : sBhi[0]);
        const __bf16* Bl   = (buf ? sBlo[1] : sBlo[0]);

        #pragma unroll
        for (int ks = 0; ks < KC; ks += 32) {
            v8bf ah0 = *(const v8bf*)(AhiR + ks + lhalf * 8);
            v8bf ah1 = *(const v8bf*)(AhiR + ks + 16 + lhalf * 8);
            v8bf al0 = *(const v8bf*)(AloR + ks + lhalf * 8);
            v8bf al1 = *(const v8bf*)(AloR + ks + 16 + lhalf * 8);
            v16bf a_hi = __builtin_shufflevector(ah0, ah1, 0,1,2,3,4,5,6,7,
                                                 8,9,10,11,12,13,14,15);
            v16bf a_lo = __builtin_shufflevector(al0, al1, 0,1,2,3,4,5,6,7,
                                                 8,9,10,11,12,13,14,15);
            #pragma unroll
            for (int s = 0; s < 2; ++s) {
                int c = nw * 32 + s * 16 + l16;
                v16bf b_hi = __builtin_shufflevector(*(const v8bf*)(Bh + c * KPAD + ks + lhalf * 8), *(const v8bf*)(Bh + c * KPAD + ks + 16 + lhalf * 8), 0,1,2,3,4,5,6,7,8,9,10,11,12,13,14,15);
                v16bf b_lo = __builtin_shufflevector(*(const v8bf*)(Bl + c * KPAD + ks + lhalf * 8), *(const v8bf*)(Bl + c * KPAD + ks + 16 + lhalf * 8), 0,1,2,3,4,5,6,7,8,9,10,11,12,13,14,15);
                v8f acc = s ? acc1 : acc0;
                acc = __builtin_amdgcn_wmma_f32_16x16x32_bf16(
                          false, a_hi, false, b_hi, (short)0, acc, false, false);
                acc = __builtin_amdgcn_wmma_f32_16x16x32_bf16(
                          false, a_lo, false, b_hi, (short)0, acc, false, false);
                acc = __builtin_amdgcn_wmma_f32_16x16x32_bf16(
                          false, a_hi, false, b_lo, (short)0, acc, false, false);
                if (s) acc1 = acc; else acc0 = acc;
            }
        }
    }

    #pragma unroll
    for (int s = 0; s < 2; ++s) {
        v8f acc = s ? acc1 : acc0;
        int n = n0 + nw * 32 + s * 16 + l16;
        float cn = c2[n];
        #pragma unroll
        for (int j = 0; j < 8; ++j) {
            int m = b0 + mw * 16 + lhalf * 8 + j;
            float d = x2[m] + cn - 2.0f * acc[j];
            d2[(size_t)m * KCODES + n] = d < 0.f ? 0.f : d;
        }
    }
}

// ---------------------------------------------------------------------------
// Kernel 2 (slow path): in-loop conversion GEMM
// ---------------------------------------------------------------------------
__global__ __launch_bounds__(256) void vq_gemm_d2(const float* __restrict__ lat,
                                                  const float* __restrict__ cb,
                                                  float* __restrict__ ws) {
    const float* x2 = ws;
    const float* c2 = ws + BATCH;
    float* d2 = ws + WS_D2_OFF;

    __shared__ __bf16 Ahi[MT * KPAD];
    __shared__ __bf16 Alo[MT * KPAD];
    __shared__ __bf16 Bhi[NT * KPAD];
    __shared__ __bf16 Blo[NT * KPAD];

    const int b0 = blockIdx.x * MT;
    const int n0 = blockIdx.y * NT;

    const int tid   = threadIdx.x;
    const int lane  = tid & 31;
    const int wv    = tid >> 5;
    const int mw    = wv >> 1;
    const int nw    = wv & 1;
    const int lhalf = lane >> 4;
    const int l16   = lane & 15;

    v8f acc0 = {};
    v8f acc1 = {};

    int   s_row[8];
    int   s_cv[8];
    const float* s_base[8];
    #pragma unroll
    for (int q = 0; q < 8; ++q) {
        int id  = tid + q * 256;
        int row = (id & 1023) >> 4;
        int cv  = id & 15;
        s_row[q]  = row;
        s_cv[q]   = cv;
        s_base[q] = (id < 1024) ? (lat + (size_t)(b0 + row) * F_DIM + cv * 4)
                                : (cb  + (size_t)(n0 + row) * F_DIM + cv * 4);
    }

    float4 pre[8];
    #pragma unroll
    for (int q = 0; q < 8; ++q) pre[q] = *(const float4*)(s_base[q]);

    const __bf16* Ahi_r = Ahi + (mw * 16 + l16) * KPAD;
    const __bf16* Alo_r = Alo + (mw * 16 + l16) * KPAD;

    for (int f0 = 0; f0 < F_DIM; f0 += KC) {
        #pragma unroll
        for (int q = 0; q < 8; ++q) {
            float4 v = pre[q];
            __bf16 h0 = (__bf16)v.x, h1 = (__bf16)v.y,
                   h2 = (__bf16)v.z, h3 = (__bf16)v.w;
            v4bf hi4 = {h0, h1, h2, h3};
            v4bf lo4 = {(__bf16)(v.x - (float)h0), (__bf16)(v.y - (float)h1),
                        (__bf16)(v.z - (float)h2), (__bf16)(v.w - (float)h3)};
            int off = s_row[q] * KPAD + s_cv[q] * 4;
            if (q < 4) { *(v4bf*)(Ahi + off) = hi4; *(v4bf*)(Alo + off) = lo4; }
            else       { *(v4bf*)(Bhi + off) = hi4; *(v4bf*)(Blo + off) = lo4; }
        }
        __syncthreads();

        int fn = f0 + KC;
        if (fn < F_DIM) {
            #pragma unroll
            for (int q = 0; q < 8; ++q) pre[q] = *(const float4*)(s_base[q] + fn);
        }

        #pragma unroll
        for (int ks = 0; ks < KC; ks += 32) {
            v8bf ah0 = *(const v8bf*)(Ahi_r + ks + lhalf * 8);
            v8bf ah1 = *(const v8bf*)(Ahi_r + ks + 16 + lhalf * 8);
            v8bf al0 = *(const v8bf*)(Alo_r + ks + lhalf * 8);
            v8bf al1 = *(const v8bf*)(Alo_r + ks + 16 + lhalf * 8);
            v16bf a_hi = __builtin_shufflevector(ah0, ah1, 0,1,2,3,4,5,6,7,
                                                 8,9,10,11,12,13,14,15);
            v16bf a_lo = __builtin_shufflevector(al0, al1, 0,1,2,3,4,5,6,7,
                                                 8,9,10,11,12,13,14,15);
            #pragma unroll
            for (int s = 0; s < 2; ++s) {
                int c = nw * 32 + s * 16 + l16;
                v16bf b_hi = __builtin_shufflevector(*(const v8bf*)(Bhi + c * KPAD + ks + lhalf * 8), *(const v8bf*)(Bhi + c * KPAD + ks + 16 + lhalf * 8), 0,1,2,3,4,5,6,7,8,9,10,11,12,13,14,15);
                v16bf b_lo = __builtin_shufflevector(*(const v8bf*)(Blo + c * KPAD + ks + lhalf * 8), *(const v8bf*)(Blo + c * KPAD + ks + 16 + lhalf * 8), 0,1,2,3,4,5,6,7,8,9,10,11,12,13,14,15);
                v8f acc = s ? acc1 : acc0;
                acc = __builtin_amdgcn_wmma_f32_16x16x32_bf16(
                          false, a_hi, false, b_hi, (short)0, acc, false, false);
                acc = __builtin_amdgcn_wmma_f32_16x16x32_bf16(
                          false, a_lo, false, b_hi, (short)0, acc, false, false);
                acc = __builtin_amdgcn_wmma_f32_16x16x32_bf16(
                          false, a_hi, false, b_lo, (short)0, acc, false, false);
                if (s) acc1 = acc; else acc0 = acc;
            }
        }
        __syncthreads();
    }

    #pragma unroll
    for (int s = 0; s < 2; ++s) {
        v8f acc = s ? acc1 : acc0;
        int n = n0 + nw * 32 + s * 16 + l16;
        float cn = c2[n];
        #pragma unroll
        for (int j = 0; j < 8; ++j) {
            int m = b0 + mw * 16 + lhalf * 8 + j;
            float d = x2[m] + cn - 2.0f * acc[j];
            d2[(size_t)m * KCODES + n] = d < 0.f ? 0.f : d;
        }
    }
}

// ---------------------------------------------------------------------------
__global__ void vq_usage_init(const float* __restrict__ in_usage,
                              float* __restrict__ out_usage) {
    int i = blockIdx.x * 256 + threadIdx.x;
    if (i < KCODES) out_usage[i] = in_usage[i];
}

__global__ __launch_bounds__(256) void vq_argmin(const float* __restrict__ ws,
                                                 float* __restrict__ out_idx,
                                                 float* __restrict__ out_mind,
                                                 float* __restrict__ out_usage,
                                                 int* __restrict__ ws_idx) {
    int r = blockIdx.x;
    const float* d2 = ws + WS_D2_OFF + (size_t)r * KCODES;
    int tid = threadIdx.x;
    float best = 3.4e38f;
    int bi = 0;
    for (int k = tid; k < KCODES; k += 256) {
        float d = d2[k];
        if (d < best) { best = d; bi = k; }
    }
    __shared__ float sv[256];
    __shared__ int   si[256];
    sv[tid] = best; si[tid] = bi;
    __syncthreads();
    for (int stride = 128; stride > 0; stride >>= 1) {
        if (tid < stride) {
            float ov = sv[tid + stride]; int oi = si[tid + stride];
            if (ov < sv[tid] || (ov == sv[tid] && oi < si[tid])) {
                sv[tid] = ov; si[tid] = oi;
            }
        }
        __syncthreads();
    }
    if (tid == 0) {
        int k = si[0];
        float d = sv[0];
        out_idx[r]  = (float)k;
        out_mind[r] = sqrtf(d < 0.f ? 0.f : d);
        ws_idx[r]   = k;
        atomicAdd(&out_usage[k], 1.0f);
    }
}

__global__ __launch_bounds__(256) void vq_gather(const float* __restrict__ cb,
                                                 const int* __restrict__ ws_idx,
                                                 float* __restrict__ outq) {
    int b = blockIdx.x;
    int k = ws_idx[b];
    const float4* src = (const float4*)(cb + (size_t)k * F_DIM);
    float4* dst = (float4*)(outq + (size_t)b * F_DIM);
    for (int i = threadIdx.x; i < F_DIM / 4; i += 256) dst[i] = src[i];
}

// ---------------------------------------------------------------------------
extern "C" void kernel_launch(void* const* d_in, const int* in_sizes, int n_in,
                              void* d_out, int out_size, void* d_ws, size_t ws_size,
                              hipStream_t stream) {
    const float* lat      = (const float*)d_in[0];
    const float* cb       = (const float*)d_in[1];
    const float* usage_in = (const float*)d_in[2];

    float* ws = (float*)d_ws;
    int*   ws_idx = (int*)(ws + WS_IDX_OFF);

    float* out_idx   = (float*)d_out;
    float* out_q     = out_idx + BATCH;
    float* out_mind  = out_q + (size_t)BATCH * F_DIM;
    float* out_usage = out_mind + BATCH;

    dim3 grid(BATCH / MT, KCODES / NT);   // 8 x 16 = 128 blocks

    if (ws_size >= WS_NEED_FAST) {
        __bf16* Lhi = (__bf16*)(ws + WS_SPLIT_OFF);
        __bf16* Llo = Lhi + (size_t)BATCH * F_DIM;
        __bf16* Chi = Llo + (size_t)BATCH * F_DIM;
        __bf16* Clo = Chi + (size_t)KCODES * F_DIM;
        vq_norms_split<<<BATCH + KCODES, 256, 0, stream>>>(lat, cb, ws,
                                                           Lhi, Llo, Chi, Clo);
        vq_gemm_d2_bf16<<<grid, 256, 0, stream>>>(Lhi, Llo, Chi, Clo, ws);
    } else {
        vq_norms<<<BATCH + KCODES, 256, 0, stream>>>(lat, cb, ws);
        vq_gemm_d2<<<grid, 256, 0, stream>>>(lat, cb, ws);
    }

    vq_usage_init<<<(KCODES + 255) / 256, 256, 0, stream>>>(usage_in, out_usage);
    vq_argmin<<<BATCH, 256, 0, stream>>>(ws, out_idx, out_mind, out_usage, ws_idx);
    vq_gather<<<BATCH, 256, 0, stream>>>(cb, ws_idx, out_q);
}
